// HSTUModel_27539330302472
// MI455X (gfx1250) — matrix operand order repaired and checked by reading, not compiled, not run on hardware
//
#include <hip/hip_runtime.h>
#include <hip/hip_bf16.h>

typedef _Float16 half_t;
typedef __attribute__((ext_vector_type(16))) _Float16 v16h;
typedef __attribute__((ext_vector_type(8)))  _Float16 v8h;
typedef __attribute__((ext_vector_type(8)))  float    v8f;

#define N_SEQ   20
#define NPAD    32
#define DH      160
#define NHEADS  4
#define ADIM    64
#define LIN     128
#define DUVQK   1536
#define ULEN    512
#define BATCH   4096
#define NSPEC   1024

// ---------------- LDS layout (bytes), total 138528 (<160KB -> 2 WGs/WGP) ---
#define SM_X     0        // float [20][160]   12800
#define SM_NXA   12800    // half  [32][160]   10240
#define SM_U     23040    // half  [20][512]   20480
#define SM_VT    43520    // half  [512][32]   32768
#define SM_Q     76288    // half  [32][256]   16384
#define SM_K     92672    // half  [32][256]   16384
#define SM_GATED 76288    // half  [32][512]   aliases Q+K (q,k dead by then)
#define SM_S     109056   // half  [4][32][32]  8192
#define SM_AV    117248   // half  [20][512]   20480
#define SM_REL   137728   // float [39] (pad to 160B)
#define SM_OB    137888   // float [160]
#define SM_TOTAL 138528

// ---------------- workspace layout (bytes), ~3.44 MB ----------------------
#define WS_UVQKT 0                                   // half [2][1536][160]
#define WS_OWT   (WS_UVQKT + 2*DUVQK*DH*2)           // half [2][160][512]
#define WS_HEADT (WS_OWT   + 2*DH*ULEN*2)            // half [3][1024][160]
#define WS_REP   (WS_HEADT + 3*NSPEC*DH*2)           // half [4096][160]

__device__ __forceinline__ float fast_rcp(float x){ return __builtin_amdgcn_rcpf(x); }
__device__ __forceinline__ float sigmf(float x)  { return fast_rcp(1.0f + __expf(-x)); }
__device__ __forceinline__ float siluf(float x)  { return x * sigmf(x); }

__device__ __forceinline__ float wave_sum(float v){
#pragma unroll
  for (int m = 16; m >= 1; m >>= 1) v += __shfl_xor(v, m, 32);
  return v;
}

// A fragment 16x32 f16, row-major source (stride in halves).
// Contiguous k order used here (same as load_b_f16): lanes0-15 row=lane K=0..15, lanes16-31 K=16..31 (16 contiguous halves).
__device__ __forceinline__ v16h load_a_f16(const half_t* base, int row0, int k0, int stride){
  const int lane = threadIdx.x & 31;
  const int r    = lane & 15;
  const int kb   = (lane < 16) ? 0 : 16;
  const half_t* p = base + (size_t)(row0 + r) * stride + k0 + kb;
  v8h lo = *(const v8h*)(p);
  v8h hi = *(const v8h*)(p + 8);
  v16h a;
#pragma unroll
  for (int i = 0; i < 8; i++){ a[i] = lo[i]; a[8+i] = hi[i]; }
  return a;
}

// B fragment 32x16 f16 from "K-contiguous" storage: element(k,n) at base+n*stride+k.
// Contiguous k order used here (same as load_a_f16): lane n<16 col=n K=0..15 (2/vgpr), lane16+n K=16..31 -> 16 contiguous halves.
__device__ __forceinline__ v16h load_b_f16(const half_t* base, int k0, int n0, int stride){
  const int lane = threadIdx.x & 31;
  const int n    = lane & 15;
  const int kb   = (lane < 16) ? 0 : 16;
  const half_t* p = base + (size_t)(n0 + n) * stride + k0 + kb;
  v8h lo = *(const v8h*)(p);
  v8h hi = *(const v8h*)(p + 8);
  v16h b;
#pragma unroll
  for (int i = 0; i < 8; i++){ b[i] = lo[i]; b[8+i] = hi[i]; }
  return b;
}

__device__ __forceinline__ v8f wmma_f32_f16(v16h a, v16h b, v8f c){
  return __builtin_amdgcn_wmma_f32_16x16x32_f16(false, a, false, b, (short)0, c, false, false);
}

// ---------------------------------------------------------------------------
// weight convert: dst[b][n][k] = (f16) src[b][k*N + n]   (transpose to K-contig)
__global__ __launch_bounds__(256)
void convert_transpose_kernel(const float* __restrict__ src, half_t* __restrict__ dst,
                              int K, int N, int batch){
  const long long total  = (long long)batch * K * N;
  const long long stride = (long long)gridDim.x * blockDim.x;
  for (long long i = (long long)blockIdx.x * blockDim.x + threadIdx.x; i < total; i += stride){
    long long bm  = i / ((long long)K * N);
    long long rem = i - bm * (long long)K * N;
    int n = (int)(rem / K);
    int k = (int)(rem - (long long)n * K);
    dst[i] = (half_t)src[bm * (long long)K * N + (long long)k * N + n];
  }
}

// ---------------------------------------------------------------------------
// fused 2-block HSTU encoder: one sample per workgroup, 8 waves.
__global__ __launch_bounds__(256)
void hstu_encode_kernel(const float* __restrict__ emb,       // [B][20][128]
                        const int*   __restrict__ delta_t,   // [B][20]
                        const int*   __restrict__ lengths,   // [B]
                        const float* __restrict__ pos_emb,   // [20][160]
                        const float* __restrict__ rating_emb,// [10][32]
                        const float* __restrict__ o_b,       // [2][160]
                        const float* __restrict__ rel_w,     // [2][39]
                        const half_t* __restrict__ uvqkT,    // [2][1536][160]
                        const half_t* __restrict__ owT,      // [2][160][512]
                        half_t* __restrict__ rep)            // [B][160]
{
  extern __shared__ char smem[];
  float*  xs   = (float*) (smem + SM_X);
  half_t* nxA  = (half_t*)(smem + SM_NXA);
  half_t* uS   = (half_t*)(smem + SM_U);
  half_t* vT   = (half_t*)(smem + SM_VT);
  half_t* qS   = (half_t*)(smem + SM_Q);
  half_t* kS   = (half_t*)(smem + SM_K);
  half_t* gate = (half_t*)(smem + SM_GATED);
  half_t* sS   = (half_t*)(smem + SM_S);
  half_t* avS  = (half_t*)(smem + SM_AV);
  float*  relS = (float*) (smem + SM_REL);
  float*  obS  = (float*) (smem + SM_OB);

  const int s      = blockIdx.x;
  const int tid    = threadIdx.x;
  // wave index is physically wave-uniform on wave32; readfirstlane makes the
  // compiler treat it as an SGPR -> all wave-derived branches become SALU.
  const int wave   = __builtin_amdgcn_readfirstlane(tid >> 5);
  const int lane   = tid & 31;
  const int colL   = lane & 15;
  const int rowOff = (lane < 16) ? 0 : 8;
  const int len    = lengths[s];

  // ---- preproc: x = concat(emb, rating[dt])*sqrt(160) + pos, masked --------
  const float scale = 12.6491106406735173f;   // sqrt(160)
  for (int i = tid; i < N_SEQ * DH; i += 256){
    int row = i / DH, col = i - row * DH;
    float v = 0.0f;
    if (row < len){
      float e = (col < 128) ? emb[((size_t)s * N_SEQ + row) * 128 + col]
                            : rating_emb[delta_t[s * N_SEQ + row] * 32 + (col - 128)];
      v = e * scale + pos_emb[row * DH + col];
    }
    xs[i] = v;
  }
  __syncthreads();

  for (int blk = 0; blk < 2; blk++){
    const half_t* Wt = uvqkT + (size_t)blk * DUVQK * DH;
    const half_t* Ot = owT   + (size_t)blk * DH * ULEN;
    const float*  rw = rel_w + blk * (2 * N_SEQ - 1);
    const float*  ob = o_b   + blk * DH;

    // ---- phase 1: LN(x) -> f16 A-staging; stage rel/o_b into LDS ----------
    for (int i = tid; i < 2 * N_SEQ - 1; i += 256) relS[i] = rw[i];
    for (int i = tid; i < DH; i += 256)            obS[i]  = ob[i];
    for (int row = wave; row < N_SEQ; row += 8){
      float a = 0.f, a2 = 0.f;
      for (int c = lane; c < DH; c += 32){ float v = xs[row*DH+c]; a += v; a2 += v*v; }
      a = wave_sum(a); a2 = wave_sum(a2);
      float mu  = a  * (1.0f / DH);
      float var = a2 * (1.0f / DH) - mu * mu;
      float rs  = rsqrtf(var + 1e-6f);
      for (int c = lane; c < DH; c += 32)
        nxA[row*DH + c] = (half_t)((xs[row*DH+c] - mu) * rs);
    }
    for (int i = tid; i < (NPAD - N_SEQ) * DH; i += 256) nxA[N_SEQ*DH + i] = (half_t)0.0f;
    __syncthreads();

    // ---- phase 2: uvqk = silu(nx @ W) : [32x160]@[160x1536] ---------------
    v16h a0c[5], a1c[5];
#pragma unroll
    for (int kc = 0; kc < 5; kc++){
      a0c[kc] = load_a_f16(nxA, 0,  kc*32, DH);
      a1c[kc] = load_a_f16(nxA, 16, kc*32, DH);
    }
    for (int t = wave; t < DUVQK/16; t += 8){     // 96 column tiles, 12/wave
      const int n0 = t * 16;                      // uniform (SGPR) per wave
      v8f acc0 = {}; v8f acc1 = {};
#pragma unroll
      for (int kc = 0; kc < 5; kc++){
        v16h bf = load_b_f16(Wt, kc*32, n0, DH);
        acc0 = wmma_f32_f16(a0c[kc], bf, acc0);
        acc1 = wmma_f32_f16(a1c[kc], bf, acc1);
      }
      // scatter: destination region is uniform per tile -> scalar branches
#pragma unroll
      for (int hm = 0; hm < 2; hm++){
        v8f acc = hm ? acc1 : acc0;
        const int rbase = hm * 16 + rowOff;
        if (n0 < ULEN){                                   // u [20][512]
#pragma unroll
          for (int r = 0; r < 8; r++){
            int row = rbase + r;
            if (hm == 0 || row < N_SEQ)
              uS[row*ULEN + n0 + colL] = (half_t)siluf(acc[r]);
          }
        } else if (n0 < 2*ULEN){                          // v -> vT [512][32]
          const int c = (n0 - ULEN) + colL;
#pragma unroll
          for (int r = 0; r < 8; r++){
            int row = rbase + r;
            float v = siluf(acc[r]);
            vT[c*NPAD + row] = (hm == 0 || row < N_SEQ) ? (half_t)v : (half_t)0.0f;
          }
        } else if (n0 < 2*ULEN + NHEADS*ADIM){            // q [32][256]
          const int c = (n0 - 2*ULEN) + colL;
#pragma unroll
          for (int r = 0; r < 8; r++){
            int row = rbase + r;
            float v = siluf(acc[r]);
            qS[row*256 + c] = (hm == 0 || row < N_SEQ) ? (half_t)v : (half_t)0.0f;
          }
        } else {                                          // k [32][256]
          const int c = (n0 - 2*ULEN - NHEADS*ADIM) + colL;
#pragma unroll
          for (int r = 0; r < 8; r++){
            int row = rbase + r;
            float v = siluf(acc[r]);
            kS[row*256 + c] = (hm == 0 || row < N_SEQ) ? (half_t)v : (half_t)0.0f;
          }
        }
      }
    }
    __syncthreads();

    // ---- phase 3: S = select(causal, silu(q k^T + relbias)/N, 0) ----------
    {
      const int h  = wave >> 1;
      const int n0 = (wave & 1) * 16;
      v8f s0 = {}; v8f s1 = {};
#pragma unroll
      for (int kc = 0; kc < 2; kc++){
        v16h bf = load_b_f16(kS + h*ADIM, kc*32, n0, 256);      // (attndim, keypos)
        v16h a0 = load_a_f16(qS, 0,  h*ADIM + kc*32, 256);
        v16h a1 = load_a_f16(qS, 16, h*ADIM + kc*32, 256);
        s0 = wmma_f32_f16(a0, bf, s0);
        s1 = wmma_f32_f16(a1, bf, s1);
      }
#pragma unroll
      for (int hm = 0; hm < 2; hm++){
        v8f acc = hm ? s1 : s0;
        const int rbase = hm * 16 + rowOff;
#pragma unroll
        for (int r = 0; r < 8; r++){
          int row = rbase + r;
          int col = n0 + colL;
          float out = 0.0f;
          if (row < N_SEQ && col <= row){
            float v = acc[r] + relS[(N_SEQ - 1) + col - row];
            out = siluf(v) * (1.0f / N_SEQ);
          }
          sS[h*NPAD*NPAD + row*NPAD + col] = (half_t)out;
        }
      }
    }
    __syncthreads();

    // ---- phase 4: av = S @ v : per head [32x32]@[32x128] ------------------
    for (int it = wave; it < 32; it += 8){
      const int h  = it >> 3;
      const int n0 = (it & 7) * 16;
      v16h bf = load_b_f16(vT + h*LIN*NPAD, 0, n0, NPAD);        // (keypos, dim)
      v16h a0 = load_a_f16(sS + h*NPAD*NPAD, 0,  0, NPAD);
      v16h a1 = load_a_f16(sS + h*NPAD*NPAD, 16, 0, NPAD);
      v8f acc0 = {}; v8f acc1 = {};
      acc0 = wmma_f32_f16(a0, bf, acc0);
      acc1 = wmma_f32_f16(a1, bf, acc1);
#pragma unroll
      for (int hm = 0; hm < 2; hm++){
        v8f acc = hm ? acc1 : acc0;
        const int rbase = hm * 16 + rowOff;
#pragma unroll
        for (int r = 0; r < 8; r++){
          int row = rbase + r;
          if (hm == 0 || row < N_SEQ)
            avS[row*ULEN + h*LIN + n0 + colL] = (half_t)acc[r];
        }
      }
    }
    __syncthreads();

    // ---- phase 5: gated = u * LayerNorm(av)  (aliases q/k LDS) ------------
    for (int row = wave; row < N_SEQ; row += 8){
      float a = 0.f, a2 = 0.f;
      for (int c = lane; c < ULEN; c += 32){ float v = (float)avS[row*ULEN+c]; a += v; a2 += v*v; }
      a = wave_sum(a); a2 = wave_sum(a2);
      float mu  = a  * (1.0f / ULEN);
      float var = a2 * (1.0f / ULEN) - mu * mu;
      float rs  = rsqrtf(var + 1e-6f);
      for (int c = lane; c < ULEN; c += 32){
        float nv = ((float)avS[row*ULEN+c] - mu) * rs;
        gate[row*ULEN + c] = (half_t)((float)uS[row*ULEN+c] * nv);
      }
    }
    for (int i = tid; i < (NPAD - N_SEQ) * ULEN; i += 256) gate[N_SEQ*ULEN + i] = (half_t)0.0f;
    __syncthreads();

    // ---- phase 6: x = (gated @ o_w + o_b + x) * valid ---------------------
    for (int t = wave; t < DH/16; t += 8){        // 10 tiles
      const int n0 = t * 16;
      v8f acc0 = {}; v8f acc1 = {};
#pragma unroll 4
      for (int kc = 0; kc < 16; kc++){
        v16h bf = load_b_f16(Ot, kc*32, n0, ULEN);
        v16h a0 = load_a_f16(gate, 0,  kc*32, ULEN);
        v16h a1 = load_a_f16(gate, 16, kc*32, ULEN);
        acc0 = wmma_f32_f16(a0, bf, acc0);
        acc1 = wmma_f32_f16(a1, bf, acc1);
      }
      const float bcol = obS[n0 + colL];
#pragma unroll
      for (int hm = 0; hm < 2; hm++){
        v8f acc = hm ? acc1 : acc0;
        const int rbase = hm * 16 + rowOff;
#pragma unroll
        for (int r = 0; r < 8; r++){
          int row = rbase + r;
          int col = n0 + colL;
          if (hm == 0 || row < N_SEQ){
            float nv = 0.0f;
            if (row < len) nv = acc[r] + bcol + xs[row*DH + col];
            xs[row*DH + col] = nv;
          }
        }
      }
    }
    __syncthreads();
  }

  // ---- finalize: rep = x[len-1] / max(||x||, 1e-6) ------------------------
  if (wave == 0){
    int row = len - 1;
    if (row < 0) row = 0;
    if (row > N_SEQ - 1) row = N_SEQ - 1;
    float a = 0.f;
    for (int c = lane; c < DH; c += 32){ float v = xs[row*DH+c]; a += v*v; }
    a = wave_sum(a);
    float inv = 1.0f / fmaxf(sqrtf(a), 1e-6f);
    for (int c = lane; c < DH; c += 32)
      rep[(size_t)s*DH + c] = (half_t)(xs[row*DH+c] * inv);
  }
}

// ---------------------------------------------------------------------------
// heads: sigmoid(rep[4096x160] @ W[160x1024] + b), batched WMMA GEMM.
__global__ __launch_bounds__(256)
void head_kernel(const half_t* __restrict__ rep,    // [4096][160]
                 const half_t* __restrict__ headT,  // [3][1024][160]
                 const float* __restrict__ b30,
                 const float* __restrict__ b60,
                 const float* __restrict__ b180,
                 float* __restrict__ out)            // [3][4096][1024]
{
  const int wave   = __builtin_amdgcn_readfirstlane((int)threadIdx.x >> 5);
  const int lane   = threadIdx.x & 31;
  const int colL   = lane & 15;
  const int rowOff = (lane < 16) ? 0 : 8;
  const int m0     = blockIdx.x * 32;
  const int head   = blockIdx.y >> 3;
  const int n0     = (blockIdx.y & 7) * 128 + wave * 16;
  const half_t* W  = headT + (size_t)head * NSPEC * DH;
  const float* bias = (head == 0) ? b30 : ((head == 1) ? b60 : b180);

  v8f acc0 = {}; v8f acc1 = {};
#pragma unroll
  for (int kc = 0; kc < 5; kc++){
    v16h bf = load_b_f16(W, kc*32, n0, DH);
    v16h a0 = load_a_f16(rep, m0,      kc*32, DH);
    v16h a1 = load_a_f16(rep, m0 + 16, kc*32, DH);
    acc0 = wmma_f32_f16(a0, bf, acc0);
    acc1 = wmma_f32_f16(a1, bf, acc1);
  }
  const int col = n0 + colL;
  const float bcol = bias[col];
  float* outh = out + (size_t)head * BATCH * NSPEC;
#pragma unroll
  for (int hm = 0; hm < 2; hm++){
    v8f acc = hm ? acc1 : acc0;
    const int rbase = m0 + hm * 16 + rowOff;
#pragma unroll
    for (int r = 0; r < 8; r++){
      int row = rbase + r;
      outh[(size_t)row * NSPEC + col] = sigmf(acc[r] + bcol);
    }
  }
}

// ---------------------------------------------------------------------------
extern "C" void kernel_launch(void* const* d_in, const int* in_sizes, int n_in,
                              void* d_out, int out_size, void* d_ws, size_t ws_size,
                              hipStream_t stream)
{
  const float* emb     = (const float*)d_in[0];
  const int*   delta_t = (const int*)  d_in[1];
  const int*   lengths = (const int*)  d_in[2];
  const float* pos_emb = (const float*)d_in[3];
  const float* rat_emb = (const float*)d_in[4];
  const float* uvqk    = (const float*)d_in[5];
  const float* o_w     = (const float*)d_in[6];
  const float* o_b     = (const float*)d_in[7];
  const float* rel_w   = (const float*)d_in[8];
  const float* h30_w   = (const float*)d_in[9];
  const float* h30_b   = (const float*)d_in[10];
  const float* h60_w   = (const float*)d_in[11];
  const float* h60_b   = (const float*)d_in[12];
  const float* h180_w  = (const float*)d_in[13];
  const float* h180_b  = (const float*)d_in[14];

  char* ws = (char*)d_ws;
  half_t* uvqkT = (half_t*)(ws + WS_UVQKT);
  half_t* owT   = (half_t*)(ws + WS_OWT);
  half_t* headT = (half_t*)(ws + WS_HEADT);
  half_t* rep   = (half_t*)(ws + WS_REP);

  // weight conversion (f32 -> f16, transposed to K-contiguous for WMMA B)
  convert_transpose_kernel<<<512, 256, 0, stream>>>(uvqk,   uvqkT,                DH,   DUVQK, 2);
  convert_transpose_kernel<<<512, 256, 0, stream>>>(o_w,    owT,                  ULEN, DH,    2);
  convert_transpose_kernel<<<512, 256, 0, stream>>>(h30_w,  headT,                DH,   NSPEC, 1);
  convert_transpose_kernel<<<512, 256, 0, stream>>>(h60_w,  headT + NSPEC*DH,     DH,   NSPEC, 1);
  convert_transpose_kernel<<<512, 256, 0, stream>>>(h180_w, headT + 2*NSPEC*DH,   DH,   NSPEC, 1);

  // fused per-sample encoder (both HSTU blocks) -> rep
  hstu_encode_kernel<<<BATCH, 256, SM_TOTAL, stream>>>(
      emb, delta_t, lengths, pos_emb, rat_emb, o_b, rel_w, uvqkT, owT, rep);

  // batched head GEMM + sigmoid
  dim3 hgrid(BATCH / 32, 24);
  head_kernel<<<hgrid, 256, 0, stream>>>(rep, headT, h30_b, h60_b, h180_b, (float*)d_out);
}
